// CoPRIMEModel_21861383537419
// MI455X (gfx1250) — compile-verified, matrix operand order checked by reading, not run on hardware
//
#include <hip/hip_runtime.h>
#include <hip/hip_bf16.h>
#include <math.h>

// ---------------- types ----------------
typedef __attribute__((ext_vector_type(16))) __bf16 v16bf;
typedef __attribute__((ext_vector_type(8)))  __bf16 v8bf;
typedef __attribute__((ext_vector_type(4)))  __bf16 v4bf;
typedef __attribute__((ext_vector_type(8)))  float  v8f;

// ---------------- problem constants ----------------
#define N_TOK 8192      // B*S
#define DDIM  1024
#define MDIM  4096
#define NEXP  8
#define TILE_T 32       // tokens per expert tile (2 x 16-row WMMA sub-tiles)
#define MC    256       // m-chunk width
#define XSTR  1032      // 1024 + 8 bf16 pad (conflict-free ds_load_b128)
#define HSTR  264       // 256 + 8 bf16 pad
#define MAX_TILES 256   // N_TOK / TILE_T worst case per expert

// ---------------- helpers ----------------
// A-fragment (16x32 bf16) per ISA layout: lane L: row m = L&15, hi = L>>4;
// elems 0..7 -> K = hi*8 + e ; elems 8..15 -> K = 16 + hi*8 + (e-8)
__device__ __forceinline__ v16bf load_a_frag(const __bf16* rowBase, int kOff) {
  v8bf lo = *(const v8bf*)(rowBase + kOff);
  v8bf hi = *(const v8bf*)(rowBase + kOff + 16);
  return __builtin_shufflevector(lo, hi, 0,1,2,3,4,5,6,7,8,9,10,11,12,13,14,15);
}

__device__ __forceinline__ v8f wmma_bf16(v16bf a, v16bf b, v8f c) {
  return __builtin_amdgcn_wmma_f32_16x16x32_bf16(false, a, false, b,
                                                 (short)0, c, false, false);
}

__device__ __forceinline__ float gelu_exact(float x) {
  return 0.5f * x * (1.0f + erff(x * 0.70710678118654752f));
}

// ---------------- 0: zero kernels ----------------
__global__ void zero_f4(float4* __restrict__ p, int n4) {
  int i = blockIdx.x * 256 + threadIdx.x;
  if (i < n4) p[i] = make_float4(0.f, 0.f, 0.f, 0.f);
}
__global__ void zero_i32(int* __restrict__ p, int n) {
  int i = blockIdx.x * 256 + threadIdx.x;
  if (i < n) p[i] = 0;
}

// ---------------- 1: gate logits  (n x E) ----------------
__global__ void gate_logits_k(const float* __restrict__ x,
                              const float* __restrict__ gw,
                              float* __restrict__ logits) {
  int gid = blockIdx.x * 256 + threadIdx.x;   // N_TOK * NEXP threads
  int t = gid >> 3, e = gid & 7;
  const float4* xp = (const float4*)(x + (size_t)t * DDIM);
  const float4* gp = (const float4*)(gw + (size_t)e * DDIM);
  float s = 0.f;
  #pragma unroll 8
  for (int i = 0; i < DDIM / 4; ++i) {
    float4 a = xp[i], b = gp[i];
    s = fmaf(a.x, b.x, s); s = fmaf(a.y, b.y, s);
    s = fmaf(a.z, b.z, s); s = fmaf(a.w, b.w, s);
  }
  logits[gid] = s;
}

// ---------------- 2: softmax + top-2 + scatter to expert lists ----------------
__global__ void router_topk_k(const float* __restrict__ logits,
                              int* __restrict__ cnt,
                              int* __restrict__ tok_list,
                              float* __restrict__ wgt_list) {
  int t = blockIdx.x * 256 + threadIdx.x;
  if (t >= N_TOK) return;
  float l[NEXP];
  float mx = -INFINITY;
  for (int e = 0; e < NEXP; ++e) { l[e] = logits[t * NEXP + e]; mx = fmaxf(mx, l[e]); }
  float s = 0.f;
  for (int e = 0; e < NEXP; ++e) { l[e] = __expf(l[e] - mx); s += l[e]; }
  float inv = 1.0f / s;
  int i0 = 0;
  for (int e = 1; e < NEXP; ++e) if (l[e] > l[i0]) i0 = e;       // lowest index wins ties
  int i1 = (i0 == 0) ? 1 : 0;
  for (int e = 0; e < NEXP; ++e) if (e != i0 && l[e] > l[i1]) i1 = e;
  float w0 = l[i0] * inv, w1 = l[i1] * inv;
  float norm = 1.0f / (w0 + w1 + 1e-9f);
  w0 *= norm; w1 *= norm;
  int p0 = atomicAdd(&cnt[i0], 1);
  tok_list[i0 * N_TOK + p0] = t; wgt_list[i0 * N_TOK + p0] = w0;
  int p1 = atomicAdd(&cnt[i1], 1);
  tok_list[i1 * N_TOK + p1] = t; wgt_list[i1 * N_TOK + p1] = w1;
}

// ---------------- 3: fp32 -> bf16 WMMA-tile-packed weights ----------------
// B-matrix (Kdim x Ndim) tiles of 32x16; packed as [e][nT][kT][lane][16 bf16]
__global__ void pack_w_k(const float* __restrict__ src, __bf16* __restrict__ dst,
                         int Kdim, int Ndim) {
  int gid  = blockIdx.x * 256 + threadIdx.x;
  int lane = gid & 31;
  int tIdx = gid >> 5;
  int kTiles = Kdim >> 5;
  int nTiles = Ndim >> 4;
  int kt = tIdx % kTiles;
  int nT = (tIdx / kTiles) % nTiles;
  int e  = tIdx / (kTiles * nTiles);
  int nl = lane & 15, hi = lane >> 4;
  const float* s = src + (size_t)e * Kdim * Ndim + (size_t)(nT * 16 + nl);
  v16bf v;
  #pragma unroll
  for (int el = 0; el < 16; ++el) {
    int kl = (el < 8) ? (hi * 8 + el) : (16 + hi * 8 + (el - 8));
    v[el] = (__bf16)s[(size_t)(kt * 32 + kl) * Ndim];
  }
  *(v16bf*)(dst + (size_t)tIdx * 512 + lane * 16) = v;
}

// ---------------- 4: grouped expert GEMM (the hot kernel) ----------------
__global__ __launch_bounds__(256)
void moe_expert_gemm_k(const float* __restrict__ x,
                       const __bf16* __restrict__ w1p,
                       const __bf16* __restrict__ w2p,
                       const float* __restrict__ b1,
                       const float* __restrict__ b2,
                       const int* __restrict__ cnt,
                       const int* __restrict__ tok_list,
                       const float* __restrict__ wgt_list,
                       float* __restrict__ out) {
  __shared__ __bf16 Xs[TILE_T * XSTR];   // 66 KB: 32 tokens x D (bf16, padded)
  __shared__ __bf16 Hs[TILE_T * HSTR];   // 17 KB: 32 tokens x MC (bf16, padded)
  __shared__ int   toks[TILE_T];
  __shared__ float wgts[TILE_T];

  const int e    = blockIdx.x >> 8;      // MAX_TILES = 256 per expert
  const int tile = blockIdx.x & (MAX_TILES - 1);
  const int ne   = cnt[e];
  if (tile * TILE_T >= ne) return;

  const int tid  = threadIdx.x;
  const int wave = tid >> 5;
  const int lane = tid & 31;
  const int nl   = lane & 15;
  const int hi   = lane >> 4;

  if (tid < TILE_T) {
    int idx = tile * TILE_T + tid;
    toks[tid] = (idx < ne) ? tok_list[e * N_TOK + idx] : -1;
    wgts[tid] = (idx < ne) ? wgt_list[e * N_TOK + idx] : 0.0f;
  }
  __syncthreads();

  // gather X tile: fp32 rows -> bf16 LDS (A-operand, row-major + pad)
  for (int c = tid; c < TILE_T * (DDIM / 4); c += 256) {
    int row = c >> 8;              // DDIM/4 = 256 float4 chunks per row
    int ch  = c & 255;
    int t   = toks[row];
    v4bf o = {(__bf16)0.f, (__bf16)0.f, (__bf16)0.f, (__bf16)0.f};
    if (t >= 0) {
      float4 v = *(const float4*)(x + (size_t)t * DDIM + ch * 4);
      o[0] = (__bf16)v.x; o[1] = (__bf16)v.y; o[2] = (__bf16)v.z; o[3] = (__bf16)v.w;
    }
    *(v4bf*)(&Xs[row * XSTR + ch * 4]) = o;
  }
  __syncthreads();

  // per-wave accumulators: two 16-token sub-tiles x 8 N-tiles (16 x v8f = 128 VGPRs)
  v8f acc0[8], acc1[8];
  #pragma unroll
  for (int i = 0; i < 8; ++i) {
    acc0[i] = (v8f){0.f,0.f,0.f,0.f,0.f,0.f,0.f,0.f};
    acc1[i] = (v8f){0.f,0.f,0.f,0.f,0.f,0.f,0.f,0.f};
  }

  const __bf16* xRow0 = Xs + nl * XSTR;          // sub-tile 0: rows 0..15
  const __bf16* xRow1 = Xs + (16 + nl) * XSTR;   // sub-tile 1: rows 16..31
  const __bf16* hRow0 = Hs + nl * HSTR;
  const __bf16* hRow1 = Hs + (16 + nl) * HSTR;

  for (int mc = 0; mc < MDIM / MC; ++mc) {
    if (mc) __syncthreads();   // Hs consumed by phase B of previous chunk

    // ---- phase A: Hs(32xMC) = gelu(X(32xD) @ W1[:, mc-chunk] + b1) ----
    #pragma unroll
    for (int i = 0; i < 2; ++i) {
      int nTg = mc * (MC / 16) + wave * 2 + i;           // global 16-col tile in M
      const __bf16* bb = w1p + ((size_t)(e * (MDIM / 16) + nTg) * (DDIM / 32)) * 512
                             + lane * 16;
      __builtin_prefetch(bb, 0, 1);
      v8f c0 = {}, c1 = {};
      v16bf bcur = *(const v16bf*)bb;                    // software pipeline: B k=0
      #pragma unroll
      for (int kt = 0; kt < DDIM / 32; ++kt) {
        v16bf bnext = bcur;
        if (kt < DDIM / 32 - 1) bnext = *(const v16bf*)(bb + (size_t)(kt + 1) * 512);
        v16bf a0 = load_a_frag(xRow0, kt * 32 + hi * 8);
        v16bf a1 = load_a_frag(xRow1, kt * 32 + hi * 8);
        c0 = wmma_bf16(a0, bcur, c0);                    // B reused by both sub-tiles
        c1 = wmma_bf16(a1, bcur, c1);
        bcur = bnext;
      }
      float bv = b1[e * MDIM + nTg * 16 + nl];
      int colL = (wave * 2 + i) * 16 + nl;
      #pragma unroll
      for (int r = 0; r < 8; ++r) {
        int rr = r + 8 * hi;                             // token row within sub-tile
        Hs[rr * HSTR + colL]        = (__bf16)gelu_exact(c0[r] + bv);
        Hs[(16 + rr) * HSTR + colL] = (__bf16)gelu_exact(c1[r] + bv);
      }
    }
    __syncthreads();

    // ---- phase B: acc(32x128 slice per wave) += H(32xMC) @ W2[chunk, slice] ----
    #pragma unroll
    for (int i = 0; i < 8; ++i) {
      int nTd = wave * 8 + i;                            // 16-col tile of D (0..63)
      const __bf16* bb = w2p + ((size_t)(e * (DDIM / 16) + nTd) * (MDIM / 32)
                                + mc * (MC / 32)) * 512 + lane * 16;
      __builtin_prefetch(bb, 0, 1);
      v8f c0 = acc0[i], c1 = acc1[i];
      v16bf bcur = *(const v16bf*)bb;
      #pragma unroll
      for (int kt = 0; kt < MC / 32; ++kt) {
        v16bf bnext = bcur;
        if (kt < MC / 32 - 1) bnext = *(const v16bf*)(bb + (size_t)(kt + 1) * 512);
        v16bf a0 = load_a_frag(hRow0, kt * 32 + hi * 8);
        v16bf a1 = load_a_frag(hRow1, kt * 32 + hi * 8);
        c0 = wmma_bf16(a0, bcur, c0);
        c1 = wmma_bf16(a1, bcur, c1);
        bcur = bnext;
      }
      acc0[i] = c0; acc1[i] = c1;
    }
  }

  // ---- epilogue: out[tok] += w_tok * (acc + b2) ----
  #pragma unroll
  for (int i = 0; i < 8; ++i) {
    int col = (wave * 8 + i) * 16 + nl;
    float bias = b2[e * DDIM + col];
    #pragma unroll
    for (int r = 0; r < 8; ++r) {
      int rr = r + 8 * hi;
      int t0 = toks[rr];
      int t1 = toks[16 + rr];
      if (t0 >= 0)
        atomicAdd(out + (size_t)t0 * DDIM + col, wgts[rr] * (acc0[i][r] + bias));
      if (t1 >= 0)
        atomicAdd(out + (size_t)t1 * DDIM + col, wgts[16 + rr] * (acc1[i][r] + bias));
    }
  }
}

// ---------------- launch ----------------
extern "C" void kernel_launch(void* const* d_in, const int* in_sizes, int n_in,
                              void* d_out, int out_size, void* d_ws, size_t ws_size,
                              hipStream_t stream) {
  const float* x      = (const float*)d_in[0];   // (B,S,D)
  const float* gate_W = (const float*)d_in[1];   // (E,D)
  const float* W1     = (const float*)d_in[2];   // (E,D,M)
  const float* b1     = (const float*)d_in[3];   // (E,M)
  const float* W2     = (const float*)d_in[4];   // (E,M,D)
  const float* b2     = (const float*)d_in[5];   // (E,D)
  float* out = (float*)d_out;

  // workspace carve-up
  char* ws = (char*)d_ws;
  const size_t W_ELEMS = (size_t)NEXP * DDIM * MDIM;        // 33.5M
  __bf16* w1p    = (__bf16*)ws;                              //  64 MB
  __bf16* w2p    = (__bf16*)(ws + W_ELEMS * 2);              //  64 MB
  size_t  off    = W_ELEMS * 4;
  float*  logits = (float*)(ws + off);        off += (size_t)N_TOK * NEXP * 4;
  int*    cnt    = (int*)(ws + off);          off += 256;    // 8 ints, padded
  int*    tokl   = (int*)(ws + off);          off += (size_t)NEXP * N_TOK * 4;
  float*  wgtl   = (float*)(ws + off);

  // 0) zero output + counters
  {
    int n4 = out_size / 4;
    zero_f4<<<(n4 + 255) / 256, 256, 0, stream>>>((float4*)out, n4);
    zero_i32<<<1, 32, 0, stream>>>(cnt, NEXP);
  }
  // 1) gate logits
  gate_logits_k<<<(N_TOK * NEXP) / 256, 256, 0, stream>>>(x, gate_W, logits);
  // 2) router top-2 + scatter
  router_topk_k<<<N_TOK / 256, 256, 0, stream>>>(logits, cnt, tokl, wgtl);
  // 3) weight repack fp32 -> tile-packed bf16
  {
    int thr1 = NEXP * (MDIM / 16) * (DDIM / 32) * 32;  // W1: K=D, N=M
    pack_w_k<<<thr1 / 256, 256, 0, stream>>>(W1, w1p, DDIM, MDIM);
    int thr2 = NEXP * (DDIM / 16) * (MDIM / 32) * 32;  // W2: K=M, N=D
    pack_w_k<<<thr2 / 256, 256, 0, stream>>>(W2, w2p, MDIM, DDIM);
  }
  // 4) grouped expert GEMM (fixed worst-case grid; tiles past count exit early)
  moe_expert_gemm_k<<<NEXP * MAX_TILES, 256, 0, stream>>>(
      x, w1p, w2p, b1, b2, cnt, tokl, wgtl, out);
}
